// Block2x2DenseL2SSM_45732811768500
// MI455X (gfx1250) — matrix operand order repaired and checked by reading, not compiled, not run on hardware
//
#include <hip/hip_runtime.h>
#include <hip/hip_bf16.h>

// ---------- types for CDNA5 WMMA ----------
typedef __attribute__((ext_vector_type(16))) __bf16 v16bf;
typedef __attribute__((ext_vector_type(8)))  __bf16 v8bf;
typedef __attribute__((ext_vector_type(4)))  __bf16 v4bf;
typedef __attribute__((ext_vector_type(2)))  __bf16 v2bf;
typedef __attribute__((ext_vector_type(8)))  float  v8f;

// ---------- problem dims ----------
constexpr int B_    = 64;
constexpr int T_    = 2048;
constexpr int DIN   = 256;
constexpr int DOUT  = 256;
constexpr int DS    = 512;     // d_state
constexpr int NP    = 256;     // n_pairs
constexpr int M_    = B_ * T_; // 131072 GEMM rows
constexpr int NCH   = 16;      // time chunks
constexpr int LCH   = T_ / NCH;// 128

// ---------- workspace layout (bytes) ----------
constexpr size_t OFF_PRM = 0;
constexpr size_t OFF_BM  = 8192;                                   // Bm bf16 [512][256]
constexpr size_t OFF_W   = OFF_BM + (size_t)DS * DIN * 2;          // W=[C|D] bf16 [256][768]
constexpr size_t OFF_UB  = OFF_W  + (size_t)DOUT * (DS + DIN) * 2; // u bf16 [M][256]
constexpr size_t OFF_BU  = OFF_UB + (size_t)M_ * DIN * 2;          // Bu bf16 [M][512]
constexpr size_t OFF_ZS  = OFF_BU + (size_t)M_ * DS * 2;           // zs bf16 [M][512]
constexpr size_t OFF_S   = OFF_ZS + (size_t)M_ * DS * 2;           // float2 [16][64][256]
constexpr size_t OFF_CR  = OFF_S  + (size_t)NCH * B_ * NP * 8;     // float2 [16][64][256]

// ---------- gfx1250 async global->LDS staging ----------
#ifndef USE_ASYNC_LDS
#define USE_ASYNC_LDS 1
#endif

__device__ __forceinline__ void stage_b128(__bf16* dst_lds, const __bf16* src_g)
{
#if USE_ASYNC_LDS
  unsigned loff = (unsigned)(unsigned long long)(const void*)dst_lds; // low 32b = LDS byte offset
  asm volatile("global_load_async_to_lds_b128 %0, %1, off"
               :: "v"(loff), "v"((unsigned long long)(const void*)src_g)
               : "memory");
#else
  v8bf tmp = *(const v8bf*)src_g;
  *(v8bf*)dst_lds = tmp;
#endif
}

__device__ __forceinline__ void stage_wait()
{
#if USE_ASYNC_LDS
  asm volatile("s_wait_asynccnt 0x0" ::: "memory");
#endif
  __syncthreads();
}

// =====================================================================
// Kernel 1: build matrices, power-iteration spectral norm, bf16 weights
// =====================================================================
__global__ __launch_bounds__(768) void k_prepare(
    const float* __restrict__ rho_raw, const float* __restrict__ theta,
    const float* __restrict__ K12, const float* __restrict__ K21,
    const float* __restrict__ K22, const float* __restrict__ lg,
    float* __restrict__ prm, __bf16* __restrict__ BmW, __bf16* __restrict__ WW)
{
  __shared__ float v[768], w[768], rcS[256], rsS[256], red[1024], bc[2];
  const int tid = threadIdx.x;

  if (tid < 256) {
    float rho = (1.0f / (1.0f + __expf(-rho_raw[tid]))) * (1.0f - 0.001f);
    float th  = theta[tid];
    rcS[tid] = rho * __cosf(th);
    rsS[tid] = rho * __sinf(th);
    red[768 + tid] = 0.0f;
  }
  v[tid] = 1.0f;
  __syncthreads();

  for (int it = 0; it < 100; ++it) {
    float x = v[tid];
    red[tid] = x * x;
    __syncthreads();
    for (int s = 512; s > 0; s >>= 1) { if (tid < s) red[tid] += red[tid + s]; __syncthreads(); }
    float nrm = sqrtf(red[0]);
    float rinv = (nrm > 0.0f) ? 1.0f / nrm : 0.0f;
    __syncthreads();
    v[tid] = x * rinv;
    __syncthreads();

    // w = K v
    float acc = 0.0f;
    if (tid < DS) {
      int p = tid >> 1;
      if ((tid & 1) == 0) acc = rcS[p] * v[tid] - rsS[p] * v[tid + 1];
      else                acc = rsS[p] * v[tid - 1] + rcS[p] * v[tid];
      const float* row = K12 + (size_t)tid * DIN;
      for (int j = 0; j < DIN; ++j) acc += row[j] * v[DS + j];
    } else {
      int o = tid - DS;
      const float* r1 = K21 + (size_t)o * DS;
      for (int j = 0; j < DS; ++j) acc += r1[j] * v[j];
      const float* r2 = K22 + (size_t)o * DIN;
      for (int j = 0; j < DIN; ++j) acc += r2[j] * v[DS + j];
    }
    w[tid] = acc;
    __syncthreads();

    // v = K^T w
    float a2 = 0.0f;
    if (tid < DS) {
      int p = tid >> 1;
      if ((tid & 1) == 0) a2 =  rcS[p] * w[tid]     + rsS[p] * w[tid + 1];
      else                a2 = -rsS[p] * w[tid - 1] + rcS[p] * w[tid];
      for (int o = 0; o < DOUT; ++o) a2 += K21[(size_t)o * DS + tid] * w[DS + o];
    } else {
      int jj = tid - DS;
      for (int r = 0; r < DS;   ++r) a2 += K12[(size_t)r * DIN + jj] * w[r];
      for (int o = 0; o < DOUT; ++o) a2 += K22[(size_t)o * DIN + jj] * w[DS + o];
    }
    __syncthreads();
    v[tid] = a2;
    __syncthreads();
  }

  { float x = v[tid]; red[tid] = x * x; __syncthreads();
    for (int s = 512; s > 0; s >>= 1) { if (tid < s) red[tid] += red[tid + s]; __syncthreads(); } }
  if (tid == 0) {
    float sigma = sqrtf(sqrtf(red[0]));       // ||K^T K v_hat||^(1/2) -> sigma
    sigma = fmaxf(sigma, 1e-5f);
    bc[0] = 1.0f / (sigma + 0.002f);
    bc[1] = __expf(lg[0]);
    prm[1025] = sigma;
  }
  __syncthreads();
  const float inv = bc[0], gamma = bc[1];

  if (tid < NP) {
    float a = rcS[tid] * inv, b = rsS[tid] * inv;
    prm[tid]        = a;
    prm[256 + tid]  = b;
    float la = a, lb = b;                     // lambda^128 by 7 squarings
    for (int i = 0; i < 7; ++i) { float na = la * la - lb * lb; lb = 2.0f * la * lb; la = na; }
    prm[512 + tid] = la;
    prm[768 + tid] = lb;
    if (tid == 0) prm[1024] = gamma;
  }
  for (int idx = tid; idx < DS * DIN; idx += 768)
    BmW[idx] = (__bf16)(gamma * inv * K12[idx]);
  for (int idx = tid; idx < DOUT * (DS + DIN); idx += 768) {
    int o = idx / (DS + DIN), k = idx - o * (DS + DIN);
    float val = (k < DS) ? inv * K21[(size_t)o * DS + k]
                         : gamma * inv * K22[(size_t)o * DIN + (k - DS)];
    WW[idx] = (__bf16)val;
  }
}

// =====================================================================
// Kernel 2: u (f32) -> bf16
// =====================================================================
__global__ __launch_bounds__(256) void k_cvt_u(const float* __restrict__ u,
                                               __bf16* __restrict__ ub, int n4)
{
  int i = blockIdx.x * 256 + threadIdx.x;
  if (i < n4) {
    float4 f = ((const float4*)u)[i];
    v4bf o; o[0] = (__bf16)f.x; o[1] = (__bf16)f.y; o[2] = (__bf16)f.z; o[3] = (__bf16)f.w;
    ((v4bf*)ub)[i] = o;
  }
}

// ---------- CDNA5 16-bit A-fragment (16x32): two 16B chunks per lane ----------
__device__ __forceinline__ v16bf load_a16x32(const __bf16* p0)
{
  v8bf lo = *(const v8bf*)p0;
  v8bf hi = *(const v8bf*)(p0 + 8);
  v16bf f;
#pragma unroll
  for (int i = 0; i < 8; ++i) { f[i] = lo[i]; f[i + 8] = hi[i]; }
  return f;
}

// =====================================================================
// Kernel 3: Bu = u @ Bm^T (bf16 out).  B panel staged to LDS (async).
// grid (M/128, 512/64), 256 thr = 8 waves; wave: 16 rows x 64 cols
// =====================================================================
constexpr int BPAD1 = DIN + 8;   // 264 elems = 528B row stride (4-bank skew, 16B aligned)

__global__ __launch_bounds__(256) void k_gemm_bu(const __bf16* __restrict__ U,
                                                 const __bf16* __restrict__ Bm,
                                                 __bf16* __restrict__ Bu)
{
  __shared__ __bf16 Bs[64 * BPAD1];            // 33 KB
  const int lane = threadIdx.x & 31, wave = threadIdx.x >> 5;
  const int row0 = blockIdx.x * 128 + wave * 16;
  const int col0 = blockIdx.y * 64;
  const int hsel = lane >> 4, sub = lane & 15;

  // stage 64x256 bf16 B panel: 2048 x 16B chunks, 8 per thread
#pragma unroll
  for (int q = threadIdx.x; q < 64 * 32; q += 256) {
    int row = q >> 5, ch = q & 31;
    stage_b128(&Bs[row * BPAD1 + ch * 8], Bm + (size_t)(col0 + row) * DIN + ch * 8);
  }
  stage_wait();

  v8f a0 = {}, a1 = {}, a2 = {}, a3 = {};
  const __bf16* ap = U + (size_t)(row0 + sub) * DIN + hsel * 16;
  const __bf16* bl = &Bs[(size_t)sub * BPAD1 + hsel * 16];

#pragma unroll
  for (int kk = 0; kk < DIN; kk += 32) {
    v16bf a  = load_a16x32(ap + kk);
    v16bf b0 = *(const v16bf*)(bl + kk);
    v16bf b1 = *(const v16bf*)(bl + 16 * BPAD1 + kk);
    v16bf b2 = *(const v16bf*)(bl + 32 * BPAD1 + kk);
    v16bf b3 = *(const v16bf*)(bl + 48 * BPAD1 + kk);
    a0 = __builtin_amdgcn_wmma_f32_16x16x32_bf16(false, a, false, b0, (short)0, a0, false, false);
    a1 = __builtin_amdgcn_wmma_f32_16x16x32_bf16(false, a, false, b1, (short)0, a1, false, false);
    a2 = __builtin_amdgcn_wmma_f32_16x16x32_bf16(false, a, false, b2, (short)0, a2, false, false);
    a3 = __builtin_amdgcn_wmma_f32_16x16x32_bf16(false, a, false, b3, (short)0, a3, false, false);
  }
  const int srow = row0 + hsel * 8, scol = col0 + sub;
#pragma unroll
  for (int r = 0; r < 8; ++r) {
    __bf16* o = Bu + (size_t)(srow + r) * DS + scol;
    o[0]  = (__bf16)a0[r];
    o[16] = (__bf16)a1[r];
    o[32] = (__bf16)a2[r];
    o[48] = (__bf16)a3[r];
  }
}

// =====================================================================
// Kernels 4-6: chunked diagonal-complex scan over T
// =====================================================================
__global__ __launch_bounds__(256) void k_scan_local(const __bf16* __restrict__ Bu,
                                                    const float* __restrict__ prm,
                                                    float2* __restrict__ S)
{
  const int p = threadIdx.x;
  const int b = blockIdx.x >> 4, c = blockIdx.x & 15;
  const float a = prm[p], bb = prm[256 + p];
  float zx = 0.f, zy = 0.f;
  const __bf16* ptr = Bu + ((size_t)b * T_ + (size_t)c * LCH) * DS + 2 * p;
  for (int t = 0; t < LCH; ++t) {
    v2bf bu = *(const v2bf*)ptr;
    float nx = a * zx - bb * zy + (float)bu[0];
    zy = bb * zx + a * zy + (float)bu[1];
    zx = nx;
    ptr += DS;
  }
  S[((size_t)c * B_ + b) * NP + p] = make_float2(zx, zy);
}

__global__ __launch_bounds__(256) void k_scan_carry(const float* __restrict__ prm,
                                                    const float2* __restrict__ S,
                                                    float2* __restrict__ Cr)
{
  const int p = threadIdx.x, b = blockIdx.x;
  const float la = prm[512 + p], lb = prm[768 + p];   // lambda^128
  float zx = 0.f, zy = 0.f;
  for (int c = 0; c < NCH; ++c) {
    size_t idx = ((size_t)c * B_ + b) * NP + p;
    Cr[idx] = make_float2(zx, zy);
    float2 s = S[idx];
    float nx = la * zx - lb * zy + s.x;
    zy = lb * zx + la * zy + s.y;
    zx = nx;
  }
}

__global__ __launch_bounds__(256) void k_scan_apply(const __bf16* __restrict__ Bu,
                                                    const float* __restrict__ prm,
                                                    const float2* __restrict__ Cr,
                                                    __bf16* __restrict__ Z)
{
  const int p = threadIdx.x;
  const int b = blockIdx.x >> 4, c = blockIdx.x & 15;
  const float a = prm[p], bb = prm[256 + p];
  float2 z0 = Cr[((size_t)c * B_ + b) * NP + p];
  float zx = z0.x, zy = z0.y;
  size_t off = ((size_t)b * T_ + (size_t)c * LCH) * DS + 2 * p;
  const __bf16* ptr = Bu + off;
  __bf16* zp = Z + off;
  for (int t = 0; t < LCH; ++t) {
    v2bf o; o[0] = (__bf16)zx; o[1] = (__bf16)zy;    // zs[t] = pre-update state
    *(v2bf*)zp = o;
    v2bf bu = *(const v2bf*)ptr;
    float nx = a * zx - bb * zy + (float)bu[0];
    zy = bb * zx + a * zy + (float)bu[1];
    zx = nx;
    ptr += DS; zp += DS;
  }
}

// =====================================================================
// Kernel 7: y = [zs | u] @ W^T (f32 out), K=768.  W panel staged to LDS.
// =====================================================================
constexpr int KW    = DS + DIN;  // 768
constexpr int BPAD2 = KW + 8;    // 776 elems = 1552B row stride (4-bank skew, 16B aligned)

__global__ __launch_bounds__(256) void k_gemm_y(const __bf16* __restrict__ Z,
                                                const __bf16* __restrict__ U,
                                                const __bf16* __restrict__ W,
                                                float* __restrict__ Y)
{
  __shared__ __bf16 Ws[64 * BPAD2];            // ~97 KB (LDS is 320 KB/WGP)
  const int lane = threadIdx.x & 31, wave = threadIdx.x >> 5;
  const int row0 = blockIdx.x * 128 + wave * 16;
  const int col0 = blockIdx.y * 64;
  const int hsel = lane >> 4, sub = lane & 15;

  // stage 64x768 bf16 W panel: 6144 x 16B chunks, 24 per thread
  for (int q = threadIdx.x; q < 64 * 96; q += 256) {
    int row = q / 96, ch = q - row * 96;
    stage_b128(&Ws[row * BPAD2 + ch * 8], W + (size_t)(col0 + row) * KW + ch * 8);
  }
  stage_wait();

  v8f a0 = {}, a1 = {}, a2 = {}, a3 = {};
  const __bf16* az = Z + (size_t)(row0 + sub) * DS  + hsel * 16;
  const __bf16* au = U + (size_t)(row0 + sub) * DIN + hsel * 16;
  const __bf16* bl = &Ws[(size_t)sub * BPAD2 + hsel * 16];

  for (int kk = 0; kk < KW; kk += 32) {
    v16bf a  = (kk < DS) ? load_a16x32(az + kk) : load_a16x32(au + (kk - DS));
    v16bf b0 = *(const v16bf*)(bl + kk);
    v16bf b1 = *(const v16bf*)(bl + 16 * BPAD2 + kk);
    v16bf b2 = *(const v16bf*)(bl + 32 * BPAD2 + kk);
    v16bf b3 = *(const v16bf*)(bl + 48 * BPAD2 + kk);
    a0 = __builtin_amdgcn_wmma_f32_16x16x32_bf16(false, a, false, b0, (short)0, a0, false, false);
    a1 = __builtin_amdgcn_wmma_f32_16x16x32_bf16(false, a, false, b1, (short)0, a1, false, false);
    a2 = __builtin_amdgcn_wmma_f32_16x16x32_bf16(false, a, false, b2, (short)0, a2, false, false);
    a3 = __builtin_amdgcn_wmma_f32_16x16x32_bf16(false, a, false, b3, (short)0, a3, false, false);
  }
  const int srow = row0 + hsel * 8, scol = col0 + sub;
#pragma unroll
  for (int r = 0; r < 8; ++r) {
    float* o = Y + (size_t)(srow + r) * DOUT + scol;
    o[0]  = a0[r];
    o[16] = a1[r];
    o[32] = a2[r];
    o[48] = a3[r];
  }
}

// =====================================================================
extern "C" void kernel_launch(void* const* d_in, const int* in_sizes, int n_in,
                              void* d_out, int out_size, void* d_ws, size_t ws_size,
                              hipStream_t stream)
{
  const float* u    = (const float*)d_in[0];
  const float* rho  = (const float*)d_in[1];
  const float* the  = (const float*)d_in[2];
  const float* K12  = (const float*)d_in[3];
  const float* K21  = (const float*)d_in[4];
  const float* K22  = (const float*)d_in[5];
  const float* lg   = (const float*)d_in[6];
  float* y = (float*)d_out;

  char* ws = (char*)d_ws;
  float*  prm = (float*)(ws + OFF_PRM);
  __bf16* BmW = (__bf16*)(ws + OFF_BM);
  __bf16* WW  = (__bf16*)(ws + OFF_W);
  __bf16* UB  = (__bf16*)(ws + OFF_UB);
  __bf16* BU  = (__bf16*)(ws + OFF_BU);
  __bf16* ZS  = (__bf16*)(ws + OFF_ZS);
  float2* S   = (float2*)(ws + OFF_S);
  float2* CR  = (float2*)(ws + OFF_CR);

  k_prepare<<<1, 768, 0, stream>>>(rho, the, K12, K21, K22, lg, prm, BmW, WW);

  const int n4 = (M_ * DIN) / 4;
  k_cvt_u<<<(n4 + 255) / 256, 256, 0, stream>>>(u, UB, n4);

  k_gemm_bu<<<dim3(M_ / 128, DS / 64), 256, 0, stream>>>(UB, BmW, BU);

  k_scan_local<<<B_ * NCH, NP, 0, stream>>>(BU, prm, S);
  k_scan_carry<<<B_, NP, 0, stream>>>(prm, S, CR);
  k_scan_apply<<<B_ * NCH, NP, 0, stream>>>(BU, prm, CR, ZS);

  k_gemm_y<<<dim3(M_ / 128, DOUT / 64), 256, 0, stream>>>(ZS, UB, WW, y);
}
